// HandwritingRNN_6124623364283
// MI455X (gfx1250) — compile-verified, matrix operand order checked by reading, not run on hardware
//
#include <hip/hip_runtime.h>
#include <hip/hip_bf16.h>
#include <stdint.h>

// ---------------------------------------------------------------------------
// Handwriting RNN (Graves) for MI455X / gfx1250: persistent per-batch-tile
// workgroups, bf16 WMMA (v_wmma_f32_16x16x32_bf16) GEMMs with explicit
// software pipelining (B/A fragments double-buffered in VGPRs so global/LDS
// loads overlap the WMMAs instead of s_wait_loadcnt 0 per group), cell state
// held in WMMA accumulator VGPRs across all 800 timesteps, LDS staging +
// ds_add_f32 atomics for the attention window, WGP-scope prefetch of the
// L2-resident weight stream.
// ---------------------------------------------------------------------------

typedef __bf16 bf16_t;
typedef __attribute__((ext_vector_type(16))) __bf16 v16bf;
typedef __attribute__((ext_vector_type(8)))  float  v8f;

#define LSTM_H 400
#define MIX    20
#define KATT   10
#define ALPH   73
#define BATCH  128
#define TSTEPS 800
#define ULEN   64
#define TB     16          // batch rows per workgroup (WMMA M)
#define NT_H   25          // hidden 16-tiles per layer (400/16)
#define XSTR   1280        // xcat staging row stride (bf16 elems)

#define K1P 480            // padded K: [window73,x3]+h400 -> 476 -> 480
#define K2P 896            // [x3,h1,win73]+h400 -> 876 -> 896
#define K3P 1280           // [x3,h1,h2,win73]+h400 -> 1276 -> 1280
#define KGP 1216           // [h1,h2,h3] -> 1200 -> 1216
#define KAP 416            // h1 -> 400 -> 416
#define KS1 (K1P/32)
#define KS2 (K2P/32)
#define KS3 (K3P/32)
#define KSG (KGP/32)
#define KSA (KAP/32)

__device__ __forceinline__ float sigf(float x) { return 1.0f / (1.0f + __expf(-x)); }

__device__ __forceinline__ v8f vzero() {
    v8f r;
#pragma unroll
    for (int i = 0; i < 8; ++i) r[i] = 0.0f;
    return r;
}

// Load 16 bf16 halfwords as two 16-byte chunks (ds_load_b128 / global_load_b128).
__device__ __forceinline__ v16bf load_frag16(const bf16_t* p0, const bf16_t* p1) {
    v16bf r;
    *reinterpret_cast<uint4*>(&r)       = *reinterpret_cast<const uint4*>(p0);
    *(reinterpret_cast<uint4*>(&r) + 1) = *reinterpret_cast<const uint4*>(p1);
    return r;
}

// A-matrix fragment (16x32 bf16, M=row=lane&15). Per ISA layout:
// lanes<16 hold K {0..7,16..23}, lanes>=16 hold K {8..15,24..31}.
__device__ __forceinline__ v16bf load_a_frag(const bf16_t* row, int hi) {
    return load_frag16(row + hi, row + 16 + hi);
}

// B-matrix fragment, pre-swizzled: lane l owns halfwords [l*16, l*16+16).
__device__ __forceinline__ v16bf load_b_frag(const bf16_t* frag, int lane) {
    const bf16_t* p = frag + lane * 16;
    return load_frag16(p, p + 8);
}

// reuse_a as a template constant so the builtin's modifier stays an ICE.
template <bool RA>
__device__ __forceinline__ v8f wmma_bf16(v16bf a, v16bf b, v8f c) {
    // (neg_a, A, neg_b, B, c_mod, C, reuse_a, reuse_b)
    return __builtin_amdgcn_wmma_f32_16x16x32_bf16(false, a, false, b, (short)0, c,
                                                   RA, false);
}

// ---------------------------------------------------------------------------
// LSTM gate GEMM + pointwise cell update. gates = xcat @ Wc^T, Wc pre-packed
// as 1KB fragments, fragment index = ntile*KSTEPS + kstep, ntile = gate*25+jt.
// Software-pipelined: fragments for step ks+1 are loaded (distinct VGPRs)
// before the 4 WMMAs of step ks issue, so global/LDS latency overlaps XDL
// work instead of forcing s_wait_loadcnt 0 in front of every WMMA.
// ---------------------------------------------------------------------------
template <int KSTEPS>
__device__ void lstm_layer(const bf16_t* __restrict__ xcat, int xstride,
                           const bf16_t* __restrict__ wf,
                           const float* __restrict__ bc,
                           v8f* cst, bf16_t* __restrict__ hout,
                           int wave, int lane) {
    const int hi = (lane & 16) ? 8 : 0;
    const bf16_t* arow = xcat + (size_t)(lane & 15) * xstride;
#pragma unroll
    for (int idx = 0; idx < 4; ++idx) {
        int jt = wave + idx * 8;
        if (jt >= NT_H) break;
        v8f ai = vzero(), af_ = vzero(), ag = vzero(), ao = vzero();
        const bf16_t* wi  = wf + (size_t)(0 * NT_H + jt) * KSTEPS * 512;
        const bf16_t* wff = wf + (size_t)(1 * NT_H + jt) * KSTEPS * 512;
        const bf16_t* wg  = wf + (size_t)(2 * NT_H + jt) * KSTEPS * 512;
        const bf16_t* wo  = wf + (size_t)(3 * NT_H + jt) * KSTEPS * 512;

        // pipeline prologue: fragments for ks = 0
        v16bf a_c  = load_a_frag(arow, hi);
        v16bf bi_c = load_b_frag(wi,  lane);
        v16bf bf_c = load_b_frag(wff, lane);
        v16bf bg_c = load_b_frag(wg,  lane);
        v16bf bo_c = load_b_frag(wo,  lane);
        __builtin_prefetch(wi + 512, 0, 3);   // WGP-scope prefetch of next frag

        for (int ks = 0; ks < KSTEPS; ++ks) {
            v16bf a_n = a_c, bi_n = bi_c, bf_n = bf_c, bg_n = bg_c, bo_n = bo_c;
            if (ks + 1 < KSTEPS) {
                const size_t off = (size_t)(ks + 1) * 512;
                a_n  = load_a_frag(arow + (ks + 1) * 32, hi);
                bi_n = load_b_frag(wi  + off, lane);
                bf_n = load_b_frag(wff + off, lane);
                bg_n = load_b_frag(wg  + off, lane);
                bo_n = load_b_frag(wo  + off, lane);
                if (ks + 2 < KSTEPS) __builtin_prefetch(wi + off + 512, 0, 3);
            }
            // 4 WMMAs sharing the same A operand -> reuse_a hint on 2..4.
            ai  = wmma_bf16<false>(a_c, bi_c, ai);
            af_ = wmma_bf16<true >(a_c, bf_c, af_);
            ag  = wmma_bf16<true >(a_c, bg_c, ag);
            ao  = wmma_bf16<true >(a_c, bo_c, ao);
            a_c = a_n; bi_c = bi_n; bf_c = bf_n; bg_c = bg_n; bo_c = bo_n;
        }

        const int col = jt * 16 + (lane & 15);
        const float bi  = bc[col];
        const float bff = bc[LSTM_H + col];
        const float bgg = bc[2 * LSTM_H + col];
        const float bo  = bc[3 * LSTM_H + col];
        v8f c = cst[idx];
#pragma unroll
        for (int v = 0; v < 8; ++v) {
            float iv = sigf(ai[v] + bi);
            float fv = sigf(af_[v] + bff);
            float gv = tanhf(ag[v] + bgg);
            float ov = sigf(ao[v] + bo);
            float cc = fv * c[v] + iv * gv;
            c[v] = cc;
            hout[(size_t)(v + hi) * LSTM_H + col] = (bf16_t)(ov * tanhf(cc));
        }
        cst[idx] = c;
    }
}

// Single-N-tile pipelined GEMM (attention projection / output head).
template <int KSTEPS>
__device__ v8f gemm_1tile(const bf16_t* __restrict__ arow_base, int xstride,
                          const bf16_t* __restrict__ wtile, int lane, int hi) {
    v8f acc = vzero();
    v16bf a_c = load_a_frag(arow_base, hi);
    v16bf b_c = load_b_frag(wtile, lane);
    for (int ks = 0; ks < KSTEPS; ++ks) {
        v16bf a_n = a_c, b_n = b_c;
        if (ks + 1 < KSTEPS) {
            a_n = load_a_frag(arow_base + (ks + 1) * 32, hi);
            b_n = load_b_frag(wtile + (size_t)(ks + 1) * 512, lane);
        }
        acc = wmma_bf16<false>(a_c, b_c, acc);
        a_c = a_n; b_c = b_n;
    }
    return acc;
}

// ---------------------------------------------------------------------------
// Weight packer: combined [Wih | Whh] rows -> bf16 B-operand fragments.
// Fragment element (lane l, pos p): n = ntile*16 + (l&15),
// k = kstep*32 + klocal, klocal mirrors the 16-bit A-matrix lane layout.
// ---------------------------------------------------------------------------
__global__ void pack_weights(const float* __restrict__ W1, const float* __restrict__ W2,
                             int rows, int K1, int K2, int Kpad, int Npad,
                             bf16_t* __restrict__ out) {
    const int ksteps = Kpad / 32;
    const size_t total = (size_t)Npad * (size_t)Kpad;
    for (size_t i = (size_t)blockIdx.x * blockDim.x + threadIdx.x; i < total;
         i += (size_t)gridDim.x * blockDim.x) {
        size_t f = i >> 9;              // fragment index (512 elems each)
        int r = (int)(i & 511);
        int l = r >> 4;                 // lane 0..31
        int p = r & 15;                 // halfword slot 0..15
        int ntile = (int)(f / ksteps);
        int kstep = (int)(f % ksteps);
        int n = ntile * 16 + (l & 15);
        int base = (l & 16) ? 8 : 0;
        int klocal = (p < 8) ? (base + p) : (16 + base + (p - 8));
        int k = kstep * 32 + klocal;
        float v = 0.0f;
        if (n < rows) {
            if (k < K1)            v = W1[(size_t)n * K1 + k];
            else if (k < K1 + K2)  v = W2[(size_t)n * K2 + (k - K1)];
        }
        out[i] = (bf16_t)v;
    }
}

__global__ void pack_bias(const float* __restrict__ bih1, const float* __restrict__ bhh1,
                          const float* __restrict__ bih2, const float* __restrict__ bhh2,
                          const float* __restrict__ bih3, const float* __restrict__ bhh3,
                          float* __restrict__ b1, float* __restrict__ b2,
                          float* __restrict__ b3) {
    int i = blockIdx.x * blockDim.x + threadIdx.x;
    if (i < 1600)      b1[i] = bih1[i] + bhh1[i];
    else if (i < 3200) { int j = i - 1600; b2[j] = bih2[j] + bhh2[j]; }
    else if (i < 4800) { int j = i - 3200; b3[j] = bih3[j] + bhh3[j]; }
}

// ---------------------------------------------------------------------------
// Main persistent kernel: 8 workgroups x 256 threads; each owns 16 batch rows
// and scans all 800 timesteps.
// ---------------------------------------------------------------------------
__global__ __launch_bounds__(256, 1) void rnn_kernel(
    const float* __restrict__ inputs, const int* __restrict__ char_seq,
    const int* __restrict__ seq_len, const float* __restrict__ bias,
    const bf16_t* __restrict__ w1f, const bf16_t* __restrict__ w2f,
    const bf16_t* __restrict__ w3f, const bf16_t* __restrict__ wgf,
    const bf16_t* __restrict__ waf,
    const float* __restrict__ b1c, const float* __restrict__ b2c,
    const float* __restrict__ b3c,
    const float* __restrict__ ba, const float* __restrict__ bg,
    float* __restrict__ out) {
    extern __shared__ char smem[];
    bf16_t* xcat = (bf16_t*)smem;                                   // 16*1280*2 = 40960 B
    bf16_t* h1   = (bf16_t*)(smem + 40960);                         // 16*400*2
    bf16_t* h2   = h1 + 16 * LSTM_H;
    bf16_t* h3   = h2 + 16 * LSTM_H;
    float*  win  = (float*)(smem + 40960 + 3 * 12800);              // 16*80
    float*  phi  = win + 16 * 80;                                   // 16*64
    float*  ap   = phi + 16 * ULEN;                                 // 16*32
    float*  kap  = ap + 16 * 32;                                    // 16*16
    float*  olds = kap + 16 * 16;                                   // 16*128
    float*  xt   = olds + 16 * 128;                                 // 16*4
    float*  pimax = xt + 16 * 4;                                    // 16
    float*  pisum = pimax + 16;                                     // 16
    int*    chs   = (int*)(pisum + 16);                             // 16*64
    int*    lns   = chs + 16 * ULEN;                                // 16
    float*  brow  = (float*)(lns + 16);                             // 16

    const int tid  = threadIdx.x;
    const int lane = tid & 31;
    const int wave = tid >> 5;
    const int b0   = blockIdx.x * TB;

    // ---- per-workgroup init ----
    for (int i = tid; i < 16 * 80; i += 256) win[i] = 0.0f;
    for (int i = tid; i < 16 * 16; i += 256) kap[i] = 0.0f;
    for (int i = tid; i < 16 * LSTM_H; i += 256) {
        h1[i] = (bf16_t)0.0f; h2[i] = (bf16_t)0.0f; h3[i] = (bf16_t)0.0f;
    }
    for (int i = tid; i < 16 * ULEN; i += 256)
        chs[i] = char_seq[(size_t)(b0 + i / ULEN) * ULEN + (i % ULEN)];
    if (tid < 16) { lns[tid] = seq_len[b0 + tid]; brow[tid] = bias[b0 + tid]; }

    v8f c1s[4], c2s[4], c3s[4];
#pragma unroll
    for (int i = 0; i < 4; ++i) { c1s[i] = vzero(); c2s[i] = vzero(); c3s[i] = vzero(); }
    __syncthreads();

    const size_t BT20 = (size_t)BATCH * TSTEPS * 20;

    for (int t = 0; t < TSTEPS; ++t) {
        // ---- x_t ----
        if (tid < 48) {
            int b = tid / 3, j = tid % 3;
            xt[b * 4 + j] = inputs[((size_t)(b0 + b) * TSTEPS + t) * 3 + j];
        }
        __syncthreads();

        // ---- stage xcat1 = [win(73), x(3), h1_prev(400)] -> 480 ----
        for (int e = tid; e < 16 * K1P; e += 256) {
            int b = e / K1P, k = e % K1P;
            float v;
            if (k < 73)        v = win[b * 80 + k];
            else if (k < 76)   v = xt[b * 4 + (k - 73)];
            else if (k < 476)  v = (float)h1[b * LSTM_H + (k - 76)];
            else               v = 0.0f;
            xcat[(size_t)b * XSTR + k] = (bf16_t)v;
        }
        __syncthreads();
        lstm_layer<KS1>(xcat, XSTR, w1f, b1c, c1s, h1, wave, lane);
        __syncthreads();

        // ---- attention projection ap = h1 @ Wa^T + ba (waves 0,1; WMMA) ----
        if (wave < 2) {
            const int hi = (lane & 16) ? 8 : 0;
            const bf16_t* arow = h1 + (size_t)(lane & 15) * LSTM_H;
            // tail K overread past 400 hits zero-padded weights
            v8f acc = gemm_1tile<KSA>(arow, LSTM_H, waf + (size_t)wave * KSA * 512,
                                      lane, hi);
            int col = wave * 16 + (lane & 15);
            float bb = (col < 3 * KATT) ? ba[col] : 0.0f;
#pragma unroll
            for (int v = 0; v < 8; ++v) ap[(size_t)(v + hi) * 32 + col] = acc[v] + bb;
        }
        __syncthreads();

        // ---- alpha/beta/kappa ----
        if (tid < 16 * KATT) {
            int b = tid / KATT, k = tid % KATT;
            float al = __expf(ap[b * 32 + k]);
            float be = __expf(ap[b * 32 + 10 + k]);
            float kp = kap[b * 16 + k] + __expf(ap[b * 32 + 20 + k]);
            kap[b * 16 + k] = kp;
            ap[b * 32 + k] = al;
            ap[b * 32 + 10 + k] = be;
        }
        __syncthreads();

        // ---- phi[b][u] + clear window ----
        for (int e = tid; e < 16 * ULEN; e += 256) {
            int b = e >> 6, u = e & 63;
            float s = 0.0f;
#pragma unroll
            for (int k = 0; k < KATT; ++k) {
                float d = kap[b * 16 + k] - (float)u;
                s += ap[b * 32 + k] * __expf(-ap[b * 32 + 10 + k] * d * d);
            }
            phi[e] = (u < lns[b]) ? s : 0.0f;
        }
        for (int e = tid; e < 16 * 80; e += 256) win[e] = 0.0f;
        __syncthreads();

        // ---- window = phi @ onehot(chars): LDS float atomic scatter ----
        for (int e = tid; e < 16 * ULEN; e += 256) {
            int b = e >> 6;
            int a = chs[e];
            atomicAdd(&win[b * 80 + a], phi[e]);
        }
        __syncthreads();

        // ---- stage xcat2 = [x, h1, win, h2_prev] -> 896 ----
        for (int e = tid; e < 16 * K2P; e += 256) {
            int b = e / K2P, k = e % K2P;
            float v;
            if (k < 3)         v = xt[b * 4 + k];
            else if (k < 403)  v = (float)h1[b * LSTM_H + (k - 3)];
            else if (k < 476)  v = win[b * 80 + (k - 403)];
            else if (k < 876)  v = (float)h2[b * LSTM_H + (k - 476)];
            else               v = 0.0f;
            xcat[(size_t)b * XSTR + k] = (bf16_t)v;
        }
        __syncthreads();
        lstm_layer<KS2>(xcat, XSTR, w2f, b2c, c2s, h2, wave, lane);
        __syncthreads();

        // ---- stage xcat3 = [x, h1, h2, win, h3_prev] -> 1280 ----
        for (int e = tid; e < 16 * K3P; e += 256) {
            int b = e / K3P, k = e % K3P;
            float v;
            if (k < 3)          v = xt[b * 4 + k];
            else if (k < 403)   v = (float)h1[b * LSTM_H + (k - 3)];
            else if (k < 803)   v = (float)h2[b * LSTM_H + (k - 403)];
            else if (k < 876)   v = win[b * 80 + (k - 803)];
            else if (k < 1276)  v = (float)h3[b * LSTM_H + (k - 876)];
            else                v = 0.0f;
            xcat[(size_t)b * XSTR + k] = (bf16_t)v;
        }
        __syncthreads();
        lstm_layer<KS3>(xcat, XSTR, w3f, b3c, c3s, h3, wave, lane);
        __syncthreads();

        // ---- stage [h1,h2,h3] -> 1216 ----
        for (int e = tid; e < 16 * KGP; e += 256) {
            int b = e / KGP, k = e % KGP;
            float v;
            if (k < 400)        v = (float)h1[b * LSTM_H + k];
            else if (k < 800)   v = (float)h2[b * LSTM_H + (k - 400)];
            else if (k < 1200)  v = (float)h3[b * LSTM_H + (k - 800)];
            else                v = 0.0f;
            xcat[(size_t)b * XSTR + k] = (bf16_t)v;
        }
        __syncthreads();

        // ---- output GEMM: each wave owns one 16-col N-tile of 128 ----
        {
            const int hi = (lane & 16) ? 8 : 0;
            const bf16_t* arow = xcat + (size_t)(lane & 15) * XSTR;
            v8f acc = gemm_1tile<KSG>(arow, XSTR, wgf + (size_t)wave * KSG * 512,
                                      lane, hi);
            int col = wave * 16 + (lane & 15);
            float bb = (col < 121) ? bg[col] : 0.0f;
#pragma unroll
            for (int v = 0; v < 8; ++v) olds[(size_t)(v + hi) * 128 + col] = acc[v] + bb;
        }
        __syncthreads();

        // ---- pi softmax statistics ----
        if (tid < 16) {
            float sc = 1.0f + brow[tid];
            float mx = -1e30f;
            for (int c = 0; c < MIX; ++c) mx = fmaxf(mx, olds[tid * 128 + c] * sc);
            float den = 0.0f;
            for (int c = 0; c < MIX; ++c) den += __expf(olds[tid * 128 + c] * sc - mx);
            pimax[tid] = mx;
            pisum[tid] = den;
        }
        __syncthreads();

        // ---- transforms + global writes (pis, sigmas, rhos, mus, es) ----
        for (int e = tid; e < 16 * 121; e += 256) {
            int b = e / 121, c = e % 121;
            size_t bt = (size_t)(b0 + b) * TSTEPS + t;
            float v = olds[b * 128 + c];
            float bb = brow[b];
            if (c < 20) {
                out[bt * 20 + c] = __expf(v * (1.0f + bb) - pimax[b]) / pisum[b];
            } else if (c < 60) {
                out[BT20 * 4 + bt * 40 + (c - 20)] = v;                 // mus
            } else if (c < 100) {
                out[BT20 + bt * 40 + (c - 60)] = __expf(v - bb);        // sigmas
            } else if (c < 120) {
                out[BT20 * 3 + bt * 20 + (c - 100)] = tanhf(v);         // rhos
            } else {
                out[BT20 * 6 + bt] = 1.0f / (1.0f + __expf(-v));        // es
            }
        }
        __syncthreads();
    }
}

// ---------------------------------------------------------------------------
// Host launcher
// ---------------------------------------------------------------------------
extern "C" void kernel_launch(void* const* d_in, const int* in_sizes, int n_in,
                              void* d_out, int out_size, void* d_ws, size_t ws_size,
                              hipStream_t stream) {
    (void)in_sizes; (void)n_in; (void)out_size; (void)ws_size;
    const float* inputs   = (const float*)d_in[0];
    const int*   char_seq = (const int*)d_in[1];
    const int*   seq_len  = (const int*)d_in[2];
    const float* bias     = (const float*)d_in[3];
    const float* Wih1 = (const float*)d_in[4];
    const float* Whh1 = (const float*)d_in[5];
    const float* bih1 = (const float*)d_in[6];
    const float* bhh1 = (const float*)d_in[7];
    const float* Wih2 = (const float*)d_in[8];
    const float* Whh2 = (const float*)d_in[9];
    const float* bih2 = (const float*)d_in[10];
    const float* bhh2 = (const float*)d_in[11];
    const float* Wih3 = (const float*)d_in[12];
    const float* Whh3 = (const float*)d_in[13];
    const float* bih3 = (const float*)d_in[14];
    const float* bhh3 = (const float*)d_in[15];
    const float* Wa   = (const float*)d_in[16];
    const float* ba   = (const float*)d_in[17];
    const float* Wg   = (const float*)d_in[18];
    const float* bg   = (const float*)d_in[19];

    // workspace carve (bf16 fragment buffers, then f32 combined biases)
    bf16_t* w1f = (bf16_t*)d_ws;
    bf16_t* w2f = w1f + (size_t)100 * KS1 * 512;
    bf16_t* w3f = w2f + (size_t)100 * KS2 * 512;
    bf16_t* wgf = w3f + (size_t)100 * KS3 * 512;
    bf16_t* waf = wgf + (size_t)8 * KSG * 512;
    float*  b1c = (float*)(waf + (size_t)2 * KSA * 512);
    float*  b2c = b1c + 4 * LSTM_H;
    float*  b3c = b2c + 4 * LSTM_H;

    dim3 blk(256);
    pack_weights<<<2048, blk, 0, stream>>>(Wih1, Whh1, 1600, 76,  LSTM_H, K1P, 1600, w1f);
    pack_weights<<<2048, blk, 0, stream>>>(Wih2, Whh2, 1600, 476, LSTM_H, K2P, 1600, w2f);
    pack_weights<<<4096, blk, 0, stream>>>(Wih3, Whh3, 1600, 876, LSTM_H, K3P, 1600, w3f);
    pack_weights<<<1024, blk, 0, stream>>>(Wg,  nullptr, 121, 1200, 0, KGP, 128, wgf);
    pack_weights<<<64,   blk, 0, stream>>>(Wa,  nullptr, 30,  400,  0, KAP, 32,  waf);
    pack_bias<<<19, blk, 0, stream>>>(bih1, bhh1, bih2, bhh2, bih3, bhh3, b1c, b2c, b3c);

    const size_t smem_bytes = 104448;
    (void)hipFuncSetAttribute((const void*)rnn_kernel,
                              hipFuncAttributeMaxDynamicSharedMemorySize,
                              (int)smem_bytes);
    rnn_kernel<<<BATCH / TB, blk, smem_bytes, stream>>>(
        inputs, char_seq, seq_len, bias, w1f, w2f, w3f, wgf, waf,
        b1c, b2c, b3c, ba, bg, (float*)d_out);
}
